// Model_16114717294667
// MI455X (gfx1250) — compile-verified, matrix operand order checked by reading, not run on hardware
//
#include <hip/hip_runtime.h>
#include <hip/hip_bf16.h>

// Problem constants (match reference)
#define BGRAPH 512
#define NPG    128
#define EPG    1024
#define DIM    128
#define EPSBN  1e-5f
#define NEGV   -10000000000.0f

#define HP 130   // bf16 tile pitch (elements) -> 65 dwords/row: bank-friendly
#define UP 129   // f32 tile pitch

typedef __attribute__((ext_vector_type(16))) __bf16 v16bf;
typedef __attribute__((ext_vector_type(8)))  float  v8f;

union FragBF { unsigned int u[8]; v16bf v; };
union FragF  { float f[8]; v8f v; };

__device__ __forceinline__ unsigned short f2bf(float f) {
  unsigned int u = __builtin_bit_cast(unsigned int, f);
  u += 0x7FFFu + ((u >> 16) & 1u);          // round-to-nearest-even
  return (unsigned short)(u >> 16);
}

// Load a 16x32 bf16 operand tile (A layout; B uses the same pattern on W^T)
// per CDNA5 ISA 7.12.2: lane = 16*khalf + m, VGPR v holds K pair
//   K = 2*(v&3) + 8*khalf + (v>=4 ? 16 : 0)
__device__ __forceinline__ v16bf load_frag(const unsigned short* sm, int row,
                                           int k0, int lane) {
  FragBF r;
  const int khalf = lane >> 4;
#pragma unroll
  for (int v = 0; v < 8; ++v) {
    const int K = 2 * (v & 3) + 8 * khalf + ((v & 4) ? 16 : 0);
    r.u[v] = *(const unsigned int*)(sm + row * HP + k0 + K);
  }
  return r.v;
}

__global__ __launch_bounds__(256) void gcn_graph_fused(
    const float* __restrict__ x,
    const int*   __restrict__ esrc, const int* __restrict__ edst,
    const float* __restrict__ Wa, const float* __restrict__ ba,
    const float* __restrict__ ga, const float* __restrict__ bea,
    const float* __restrict__ rma, const float* __restrict__ rva,
    const float* __restrict__ Wb, const float* __restrict__ bb,
    const float* __restrict__ gb, const float* __restrict__ beb,
    const float* __restrict__ rmb, const float* __restrict__ rvb,
    const float* __restrict__ Wc, const float* __restrict__ bc,
    float* __restrict__ out) {
  extern __shared__ char smem[];
  float*          sU = (float*)smem;                       // 128*129 f32 (post-GEMM, dis-scaled)
  unsigned short* sH = (unsigned short*)(sU + 128 * UP);   // 128*130 bf16 (activations)
  unsigned short* sW = sH + 128 * HP;                      // 128*130 bf16 (W transposed)

  __shared__ int            cnt[128];
  __shared__ int            offs[129];
  __shared__ unsigned short cols[1024];
  __shared__ float          dis[128];
  __shared__ int            conn[128];
  __shared__ float          pbias[128], pscale[128], pshift[128];
  __shared__ float          pooled[128];
  __shared__ int            minc;

  const int g    = blockIdx.x;
  const int tid  = threadIdx.x;
  const int lane = tid & 31;
  const int wave = tid >> 5;

  if (tid < 128) { cnt[tid] = 0; conn[tid] = 0; pooled[tid] = 0.0f; }
  __syncthreads();

  // ---- edge pass 1: in-degree + connectivity (edges are contiguous per graph)
  for (int e = tid; e < EPG; e += 256) {
    const int s = esrc[g * EPG + e] & (NPG - 1);
    const int d = edst[g * EPG + e] & (NPG - 1);
    atomicAdd(&cnt[d], 1);
    conn[s] = 1; conn[d] = 1;
  }
  __syncthreads();

  if (tid < 128) dis[tid] = rsqrtf((float)cnt[tid] + 1.0f);  // self-loop: deg+1
  if (tid == 0) {
    int acc = 0;
    for (int n = 0; n < 128; ++n) { offs[n] = acc; acc += cnt[n]; }
    offs[128] = acc;
    int m = 0; while (m < 128 && !conn[m]) ++m;
    minc = m;
  }
  __syncthreads();
  if (tid < 128) cnt[tid] = offs[tid];  // reuse as fill cursor
  __syncthreads();

  // ---- edge pass 2: CSR column fill (incoming neighbors per node)
  for (int e = tid; e < EPG; e += 256) {
    const int s = esrc[g * EPG + e] & (NPG - 1);
    const int d = edst[g * EPG + e] & (NPG - 1);
    const int p = atomicAdd(&cnt[d], 1);
    cols[p] = (unsigned short)s;
  }

  // ---- load X tile -> bf16 LDS
  for (int i = tid * 4; i < NPG * DIM; i += 256 * 4) {
    const float4 v = *(const float4*)(x + (size_t)g * NPG * DIM + i);
    unsigned short* p = sH + (i >> 7) * HP + (i & 127);
    p[0] = f2bf(v.x); p[1] = f2bf(v.y); p[2] = f2bf(v.z); p[3] = f2bf(v.w);
  }
  __syncthreads();

  const float* Ws[3]  = {Wa, Wb, Wc};
  const float* bs[3]  = {ba, bb, bc};
  const float* gms[2] = {ga, gb};
  const float* bts[2] = {bea, beb};
  const float* rms[2] = {rma, rmb};
  const float* rvs[2] = {rva, rvb};

  for (int l = 0; l < 3; ++l) {
    // ---- stage W^T -> bf16 LDS (coalesced global read, transposed write)
    const float* W = Ws[l];
    for (int i = tid * 4; i < DIM * DIM; i += 256 * 4) {
      const float4 v = *(const float4*)(W + i);
      const int k = i >> 7, n = i & 127;
      sW[(n + 0) * HP + k] = f2bf(v.x);
      sW[(n + 1) * HP + k] = f2bf(v.y);
      sW[(n + 2) * HP + k] = f2bf(v.z);
      sW[(n + 3) * HP + k] = f2bf(v.w);
    }
    if (tid < 128) {
      pbias[tid] = bs[l][tid];
      if (l < 2) {
        const float s = gms[l][tid] * rsqrtf(rvs[l][tid] + EPSBN);
        pscale[tid] = s;
        pshift[tid] = bts[l][tid] - rms[l][tid] * s;
      }
    }
    if (l < 2) __builtin_prefetch(Ws[l + 1], 0, 0);  // global_prefetch next weights
    __syncthreads();

    // ---- GEMM: wave w owns M-strip [16w,16w+16), all 8 N-tiles, K in 4 steps
    v8f acc[8];
#pragma unroll
    for (int nt = 0; nt < 8; ++nt) acc[nt] = (v8f){0,0,0,0,0,0,0,0};
    const int mrow = wave * 16 + (lane & 15);
#pragma unroll
    for (int kt = 0; kt < 4; ++kt) {
      const v16bf a = load_frag(sH, mrow, kt * 32, lane);
#pragma unroll
      for (int nt = 0; nt < 8; ++nt) {
        const v16bf b = load_frag(sW, nt * 16 + (lane & 15), kt * 32, lane);
        acc[nt] = __builtin_amdgcn_wmma_f32_16x16x32_bf16(
            false, a, false, b, (short)0, acc[nt], false, false);
      }
    }
    // ---- epilogue: sU[M][N] = dis[M] * (H@W)[M][N]   (C/D layout 7.12.2)
    {
      const int khalf = lane >> 4;
      const int ncol  = lane & 15;
#pragma unroll
      for (int nt = 0; nt < 8; ++nt) {
        FragF d; d.v = acc[nt];
#pragma unroll
        for (int v = 0; v < 8; ++v) {
          const int M = wave * 16 + v + 8 * khalf;
          sU[M * UP + nt * 16 + ncol] = dis[M] * d.f[v];
        }
      }
    }
    __syncthreads();

    // ---- aggregation (CSR gather, no atomics) + bias/BN/ReLU
    for (int it = 0; it < 64; ++it) {
      const int n = it * 2 + (tid >> 7);
      const int f = tid & 127;
      float sum = sU[n * UP + f];                       // self-loop term
      const int e1 = offs[n + 1];
      for (int e = offs[n]; e < e1; ++e) sum += sU[cols[e] * UP + f];
      float val = dis[n] * sum + pbias[f];
      if (l < 2) {
        val = fmaxf(val * pscale[f] + pshift[f], 0.0f);  // BN + ReLU
        sH[n * HP + f] = f2bf(val);                      // next layer input
      } else {
        atomicAdd(&pooled[f], val);                      // fuse mean-pool
      }
    }
    __syncthreads();
  }

  // ---- mask + write pooled mean
  if (tid < 128) {
    const bool m = (conn[tid] != 0) && (tid != minc);
    out[(size_t)g * NPG + tid] = m ? NEGV : pooled[tid] * (1.0f / NPG);
  }
}

extern "C" void kernel_launch(void* const* d_in, const int* in_sizes, int n_in,
                              void* d_out, int out_size, void* d_ws, size_t ws_size,
                              hipStream_t stream) {
  (void)in_sizes; (void)n_in; (void)out_size; (void)d_ws; (void)ws_size;
  const float* x    = (const float*)d_in[0];
  const int*   eidx = (const int*)d_in[1];
  // d_in[2] = batch (implied by layout, unused)
  const float* W1 = (const float*)d_in[3];
  const float* b1 = (const float*)d_in[4];
  const float* g1 = (const float*)d_in[5];
  const float* be1 = (const float*)d_in[6];
  const float* rm1 = (const float*)d_in[7];
  const float* rv1 = (const float*)d_in[8];
  const float* W2 = (const float*)d_in[9];
  const float* b2 = (const float*)d_in[10];
  const float* g2 = (const float*)d_in[11];
  const float* be2 = (const float*)d_in[12];
  const float* rm2 = (const float*)d_in[13];
  const float* rv2 = (const float*)d_in[14];
  const float* W3 = (const float*)d_in[15];
  const float* b3 = (const float*)d_in[16];

  const int E = BGRAPH * EPG;
  const int* esrc = eidx;
  const int* edst = eidx + E;

  const size_t smem = (size_t)(128 * UP) * 4 + 2 * (size_t)(128 * HP) * 2; // ~131 KB
  gcn_graph_fused<<<BGRAPH, 256, smem, stream>>>(
      x, esrc, edst,
      W1, b1, g1, be1, rm1, rv1,
      W2, b2, g2, be2, rm2, rv2,
      W3, b3, (float*)d_out);
}
